// ScaledDotProductAttention_7052336300243
// MI455X (gfx1250) — matrix operand order repaired and checked by reading, not compiled, not run on hardware
//
#include <hip/hip_runtime.h>
#include <math.h>

// Problem constants (reference: H, LQ, LK, DK, DV = 16, 2048, 2048, 64, 64)
#define H_    16
#define LQ_   2048
#define LK_   2048
#define DK_   64
#define DV_   64
#define QTILE 16
#define SROW  (LK_ + 4)          // padded LDS row stride (floats) -> conflict-free row gathers
#define NWAVE 8
#define NTHREADS (NWAVE * 32)

typedef __attribute__((ext_vector_type(4)))  float    v4f;
typedef __attribute__((ext_vector_type(8)))  float    v8f;
typedef __attribute__((ext_vector_type(16))) _Float16 v16h;

static __device__ __forceinline__ v16h cvt_v16h(v8f a, v8f b) {
  v16h r;
#pragma unroll
  for (int i = 0; i < 8; ++i) { r[i] = (_Float16)a[i]; r[i + 8] = (_Float16)b[i]; }
  return r;
}

static __device__ __forceinline__ float wave_max(float x) {
#pragma unroll
  for (int off = 16; off > 0; off >>= 1) x = fmaxf(x, __shfl_xor(x, off, 32));
  return x;
}
static __device__ __forceinline__ float wave_sum(float x) {
#pragma unroll
  for (int off = 16; off > 0; off >>= 1) x += __shfl_xor(x, off, 32);
  return x;
}

extern __shared__ unsigned char smem_raw[];

__global__ __launch_bounds__(NTHREADS, 2)
void sdpa_wmma_kernel(const float* __restrict__ q,
                      const float* __restrict__ k,
                      const float* __restrict__ v,
                      const unsigned char* __restrict__ mask,
                      float* __restrict__ out,
                      float* __restrict__ attn)
{
  float* s_sc  = (float*)smem_raw;           // [QTILE][SROW] score / exp tile (~128KB)
  float* s_acc = s_sc + QTILE * SROW;        // [QTILE][DV_] output accumulator
  float* s_rs  = s_acc + QTILE * DV_;        // [QTILE] per-row 1/sum(exp)

  const int h    = blockIdx.x / (LQ_ / QTILE);
  const int qb   = blockIdx.x % (LQ_ / QTILE);
  const int tid  = threadIdx.x;
  const int wave = tid >> 5;
  const int lane = tid & 31;
  const int lh   = lane >> 4;   // lane half: selects K-subrange of A/B fragments
  const int lm   = lane & 15;

  const size_t qrow0 = (size_t)h * LQ_ + (size_t)qb * QTILE;

  // ---------------- Q A-fragments (DK=64 -> two K=32 chunks), f32 -> f16 ----------------
  // A 16x32 fragment, contiguous k order used here (same as B): lane l holds row M=l%16; elem j <-> K = (l/16)*16 + j.
  const float* qrow = q + (qrow0 + lm) * DK_;
  v16h aq[2];
#pragma unroll
  for (int c = 0; c < 2; ++c) {
    const float* p = qrow + 32 * c + lh * 16;
    v8f f0 = *(const v8f*)(p);
    v8f f1 = *(const v8f*)(p + 8);
    aq[c] = cvt_v16h(f0, f1);
  }

  // ---------------- pass 1: S = (Q K^T) / sqrt(dk), masked -> LDS ----------------
  for (int t = wave; t < LK_ / 16; t += NWAVE) {
    const float* krow = k + ((size_t)h * LK_ + (size_t)t * 16 + lm) * DK_;
    v8f acc = {};
#pragma unroll
    for (int c = 0; c < 2; ++c) {
      // B 32x16 fragment, contiguous k order used here (same as A): lane l holds col N=l%16; elem j <-> K = (l/16)*16 + j.
      const float* p = krow + 32 * c + lh * 16;
      v8f f0 = *(const v8f*)(p);
      v8f f1 = *(const v8f*)(p + 8);
      v16h b = cvt_v16h(f0, f1);
      acc = __builtin_amdgcn_wmma_f32_16x16x32_f16(false, aq[c], false, b,
                                                   (short)0, acc, false, false);
    }
    const int col = t * 16 + lm;
#pragma unroll
    for (int r = 0; r < 8; ++r) {
      // C 16x16 f32 layout: VGPR r, lanes 0-15 -> M=r, lanes 16-31 -> M=r+8
      const int row = r + lh * 8;
      const size_t midx = (qrow0 + row) * LK_ + col;
      float s = mask[midx] ? -__builtin_inff() : acc[r] * 0.125f;
      s_sc[row * SROW + col] = s;
    }
  }
  __syncthreads();

  // ---------------- softmax per row (vectorized b128 sweeps) ----------------
  // LDS keeps UNNORMALIZED exp(s - max); 1/sum folded into pass 2 via s_rs.
  // attn output gets the normalized probabilities (global_store_b128, coalesced).
#pragma unroll
  for (int rr = 0; rr < 2; ++rr) {
    const int row = wave * 2 + rr;
    float* srow = s_sc + row * SROW;

    float mx = -__builtin_inff();
    for (int i = lane * 4; i < LK_; i += 128) {
      v4f x = *(const v4f*)(srow + i);
      mx = fmaxf(mx, fmaxf(fmaxf(x[0], x[1]), fmaxf(x[2], x[3])));
    }
    mx = wave_max(mx);

    float sum = 0.f;
    for (int i = lane * 4; i < LK_; i += 128) {
      v4f x = *(const v4f*)(srow + i);
      v4f e;
#pragma unroll
      for (int j = 0; j < 4; ++j) e[j] = __expf(x[j] - mx);
      *(v4f*)(srow + i) = e;
      sum += (e[0] + e[1]) + (e[2] + e[3]);
    }
    sum = wave_sum(sum);
    const float rs = 1.f / sum;
    if (lane == 0) s_rs[row] = rs;

    float* arow = attn + (qrow0 + row) * LK_;
    for (int i = lane * 4; i < LK_; i += 128) {
      v4f e = *(const v4f*)(srow + i);
      v4f p;
#pragma unroll
      for (int j = 0; j < 4; ++j) p[j] = e[j] * rs;
      *(v4f*)(arow + i) = p;           // global_store_b128
    }
  }

  for (int i = tid; i < QTILE * DV_; i += NTHREADS) s_acc[i] = 0.f;
  __syncthreads();

  // ---------------- pass 2: O = (E V) * rs  (waves split: DV tile x K half) ----------------
  const int nt = wave & 3;     // which 16-col tile of DV
  const int kh = wave >> 2;    // 0/1 k-half
  v8f oc = {};
  for (int ch = 0; ch < LK_ / (2 * 32); ++ch) {
    const int kc = kh * (LK_ / 2) + ch * 32;
    // A fragment: rows of E from LDS (padded stride -> distinct banks per row)
    const float* pr = s_sc + lm * SROW + kc + lh * 16;
    v4f p0 = *(const v4f*)(pr);
    v4f p1 = *(const v4f*)(pr + 4);
    v4f p2 = *(const v4f*)(pr + 8);
    v4f p3 = *(const v4f*)(pr + 12);
    v16h a;
#pragma unroll
    for (int i = 0; i < 4; ++i) {
      a[i]      = (_Float16)p0[i];
      a[4 + i]  = (_Float16)p1[i];
      a[8 + i]  = (_Float16)p2[i];
      a[12 + i] = (_Float16)p3[i];
    }
    // B fragment: V rows; per-j load is lane-contiguous (coalesced), V stays L2-resident
    const float* vp = v + (size_t)(kc + lh * 16) * DV_ + nt * 16 + lm;
    v16h b;
#pragma unroll
    for (int j = 0; j < 16; ++j) b[j] = (_Float16)vp[(size_t)j * DV_];
    oc = __builtin_amdgcn_wmma_f32_16x16x32_f16(false, a, false, b,
                                                (short)0, oc, false, false);
  }
#pragma unroll
  for (int r = 0; r < 8; ++r) {
    const int row = r + lh * 8;
    const float rs = s_rs[row];                              // normalization folded here
    atomicAdd(&s_acc[row * DV_ + nt * 16 + lm], oc[r] * rs); // ds_add_f32 cross-wave reduce
  }
  __syncthreads();

  for (int i = tid; i < QTILE * DV_; i += NTHREADS) {
    out[(qrow0 + (i >> 6)) * DV_ + (i & 63)] = s_acc[i];     // coalesced O store
  }
}

extern "C" void kernel_launch(void* const* d_in, const int* in_sizes, int n_in,
                              void* d_out, int out_size, void* d_ws, size_t ws_size,
                              hipStream_t stream) {
  (void)in_sizes; (void)n_in; (void)out_size; (void)d_ws; (void)ws_size;
  const float*         q    = (const float*)d_in[0];
  const float*         k    = (const float*)d_in[1];
  const float*         v    = (const float*)d_in[2];
  const unsigned char* mask = (const unsigned char*)d_in[3];   // jax bool -> 1 byte
  // d_in[4] = train_attn: affects only gradients; forward result identical.

  float* out  = (float*)d_out;                       // outputs [H, LQ, DV]
  float* attn = out + (size_t)H_ * LQ_ * DV_;        // attn    [H, LQ, LK]

  const size_t smem =
      (size_t)(QTILE * SROW + QTILE * DV_ + QTILE) * sizeof(float); // ~135 KB
  dim3 grid(H_ * (LQ_ / QTILE));
  dim3 block(NTHREADS);
  sdpa_wmma_kernel<<<grid, block, smem, stream>>>(q, k, v, mask, out, attn);
}
